// TSPNNet_Pointer_15539191677701
// MI455X (gfx1250) — matrix operand order repaired and checked by reading, not compiled, not run on hardware
//
#include <hip/hip_runtime.h>
#include <math.h>

typedef __attribute__((ext_vector_type(16))) __bf16 v16bf;
typedef __attribute__((ext_vector_type(8)))  __bf16 v8bf;
typedef __attribute__((ext_vector_type(8)))  float  v8f;

#define B_   64
#define N_   1024
#define H_   128
#define ROWT 64      // node rows per LDS tile
#define LDA  136     // padded LDS row stride in bf16 elements (272B: breaks bank conflicts)

// Branch-free tanh: tanh(x) = copysign((1-e)/(1+e), x), e = exp(-2|x|).
// Lowers to v_exp_f32 + v_rcp_f32 + a few VALU ops; no exec-mask divergence.
__device__ __forceinline__ float fast_tanhf(float x) {
  float a = __builtin_fabsf(x);
  float e = __expf(-2.0f * a);
  float t = (1.0f - e) * __builtin_amdgcn_rcpf(1.0f + e);
  return __builtin_copysignf(t, x);
}

__device__ __forceinline__ float fast_sigmoidf(float x) {
  return __builtin_amdgcn_rcpf(1.0f + __expf(-x));
}

// ---------------------------------------------------------------------------
// Kernel 1: LSTMCell with zero (h,c) and batch-broadcast query -> hx[128], v.
// Batch-invariant, so computed exactly once. ws[0..127]=hx, ws[128]=v scalar.
// ---------------------------------------------------------------------------
__global__ __launch_bounds__(512) void lstm_hx_kernel(
    const float* __restrict__ W_ih, const float* __restrict__ b_ih,
    const float* __restrict__ b_hh, const float* __restrict__ query,
    const float* __restrict__ W_val, const float* __restrict__ b_val,
    float* __restrict__ ws) {
  __shared__ float gates[4 * H_];
  const int t = threadIdx.x;                 // 512 threads: one gate each
  const float* wr = W_ih + t * H_;
  float g = b_ih[t] + b_hh[t];
#pragma unroll 8
  for (int h = 0; h < H_; ++h) g += wr[h] * query[h];
  gates[t] = g;
  __syncthreads();
  if (t < H_) {
    float ig = gates[t];                     // torch gate order: i, f, g, o
    float gg = gates[2 * H_ + t];
    float og = gates[3 * H_ + t];
    float cx = fast_sigmoidf(ig) * fast_tanhf(gg);   // f-gate * c0 == 0
    float hx = fast_sigmoidf(og) * fast_tanhf(cx);
    ws[t] = hx;
    gates[t] = hx * W_val[t];                // reuse LDS for v-dot partials
  }
  __syncthreads();
  if (t == 0) {
    float s = b_val[0];
    for (int i = 0; i < H_; ++i) s += gates[i];
    ws[H_] = s;
  }
}

// ---------------------------------------------------------------------------
// Kernel 2: fused per-batch  emb = relu(nodes @ W_emb^T + b_emb)
//   logits = v_param . tanh(emb @ W_attn^T + hx)   (WMMA bf16, f32 accum)
//   pi     = softmax_N(logits)
// One workgroup (256 thr = 8 waves) per batch; softmax stays in LDS.
// ---------------------------------------------------------------------------
__global__ __launch_bounds__(256) void attn_kernel(
    const float* __restrict__ nf,       // [B,N,4]
    const float* __restrict__ W_emb,    // [128,4]
    const float* __restrict__ b_emb,    // [128]
    const float* __restrict__ W_attn,   // [128,128]
    const float* __restrict__ v_param,  // [128]
    const float* __restrict__ ws,       // hx[128], v at [128]
    float* __restrict__ out) {          // pi [B,N] then v [B]
  __shared__ __bf16 sW[H_][LDA];        // W_attn (row k, col h), bf16
  __shared__ __bf16 sA[ROWT][LDA];      // embedding tile (row n, col h), bf16
  __shared__ float  sLog[N_];
  __shared__ float  sHx[H_];
  __shared__ float  sV[H_];
  __shared__ float4 sNF[ROWT];
  __shared__ float  sRed[16];

  const int tid = threadIdx.x;
  const int b   = blockIdx.x;

  // stage W_attn -> bf16 LDS; hx; v_param; zero logits
  for (int i = tid; i < H_ * H_; i += 256)
    sW[i >> 7][i & (H_ - 1)] = (__bf16)W_attn[i];
  if (tid < H_) { sHx[tid] = ws[tid]; sV[tid] = v_param[tid]; }
  for (int i = tid; i < N_; i += 256) sLog[i] = 0.0f;

  // per-thread embedding column (F=4 -> one float4 of weights per h)
  const int    hcol = tid & (H_ - 1);
  const float4 we   = ((const float4*)W_emb)[hcol];
  const float  be   = b_emb[hcol];

  const int w    = tid >> 5;        // wave 0..7
  const int lane = tid & 31;
  const int lx   = lane & 15;       // A: row M ; B/C: col N
  const int hk   = lane >> 4;       // half-wave select
  const int m0   = (w >> 1) * 16;   // 16-row strip inside tile (4 strips)
  const int ctB  = (w & 1) * 4;     // column-tile half (2 halves of 4 tiles)

  __syncthreads();

  for (int t0 = 0; t0 < N_; t0 += ROWT) {
    if (tid < ROWT) sNF[tid] = ((const float4*)nf)[(size_t)b * N_ + t0 + tid];
    __syncthreads();

    // embedding tile: relu(nf . W_emb[h] + b_emb[h]) -> bf16 LDS
    for (int nl = (tid >> 7); nl < ROWT; nl += 2) {
      float4 x = sNF[nl];
      float  v = be + x.x * we.x + x.y * we.y + x.z * we.z + x.w * we.w;
      sA[nl][hcol] = (__bf16)fmaxf(v, 0.0f);
    }
    __syncthreads();

    // A fragments for this wave's 16-row strip: K(h) = (i>>3)*16 + hk*8 + (i&7)
    v16bf afr[4];
#pragma unroll
    for (int hb = 0; hb < 4; ++hb) {
      const int h0 = hb * 32;
      v8bf lo = *(const v8bf*)&sA[m0 + lx][h0 + hk * 8];
      v8bf hi = *(const v8bf*)&sA[m0 + lx][h0 + 16 + hk * 8];
      afr[hb] = __builtin_shufflevector(lo, hi, 0,1,2,3,4,5,6,7,8,9,10,11,12,13,14,15);
    }

    float logAcc[8];
#pragma unroll
    for (int r = 0; r < 8; ++r) logAcc[r] = 0.0f;

#pragma unroll
    for (int j = 0; j < 4; ++j) {          // 4 output col-tiles per wave
      const int k0 = (ctB + j) * 16;
      v8f c = {};
#pragma unroll
      for (int hb = 0; hb < 4; ++hb) {     // K = 128 in 4 chunks of 32
        const int h0 = hb * 32;
        // B fragment (32x16): lane holds col N=lx, K = (i>>3)*16 + hk*8 + (i&7)
        v8bf lo = *(const v8bf*)&sW[k0 + lx][h0 + hk * 8];
        v8bf hi = *(const v8bf*)&sW[k0 + lx][h0 + 16 + hk * 8];
        v16bf bfr = __builtin_shufflevector(lo, hi, 0,1,2,3,4,5,6,7,8,9,10,11,12,13,14,15);
        c = __builtin_amdgcn_wmma_f32_16x16x32_bf16(
                false, afr[hb], false, bfr, (short)0, c, false, false);
      }
      // fuse: tanh(Wemb + hx[k]) * v_param[k], accumulate per (row r)
      const int   kcol = k0 + lx;
      const float hx = sHx[kcol];
      const float vp = sV[kcol];
#pragma unroll
      for (int r = 0; r < 8; ++r) logAcc[r] += fast_tanhf(c[r] + hx) * vp;
    }

    // reduce each row's partial over the 16 lanes holding its 16 columns
#pragma unroll
    for (int r = 0; r < 8; ++r) {
      float v = logAcc[r];
      v += __shfl_xor(v, 1, 16);
      v += __shfl_xor(v, 2, 16);
      v += __shfl_xor(v, 4, 16);
      v += __shfl_xor(v, 8, 16);
      if (lx == 0) atomicAdd(&sLog[t0 + m0 + hk * 8 + r], v);  // ds_add_f32
    }
    __syncthreads();
  }

  // ---- softmax over sLog[0..N) within the block ----
  float m = -1e30f;
  for (int i = tid; i < N_; i += 256) m = fmaxf(m, sLog[i]);
#pragma unroll
  for (int off = 16; off >= 1; off >>= 1) m = fmaxf(m, __shfl_xor(m, off, 32));
  if (lane == 0) sRed[w] = m;
  __syncthreads();
  float bmax = sRed[0];
#pragma unroll
  for (int i = 1; i < 8; ++i) bmax = fmaxf(bmax, sRed[i]);

  float s = 0.0f;
  for (int i = tid; i < N_; i += 256) {
    float e = __expf(sLog[i] - bmax);
    sLog[i] = e;
    s += e;
  }
#pragma unroll
  for (int off = 16; off >= 1; off >>= 1) s += __shfl_xor(s, off, 32);
  if (lane == 0) sRed[8 + w] = s;
  __syncthreads();
  float bs = 0.0f;
#pragma unroll
  for (int i = 0; i < 8; ++i) bs += sRed[8 + i];
  const float inv = 1.0f / bs;
  for (int i = tid; i < N_; i += 256) out[(size_t)b * N_ + i] = sLog[i] * inv;

  if (tid == 0) out[B_ * N_ + b] = ws[H_];   // v scalar broadcast to [B,1]
}

// ---------------------------------------------------------------------------
extern "C" void kernel_launch(void* const* d_in, const int* in_sizes, int n_in,
                              void* d_out, int out_size, void* d_ws, size_t ws_size,
                              hipStream_t stream) {
  const float* node_features = (const float*)d_in[0];
  // d_in[1] adjacency_matrix: UNUSED by the reference -> never read (256 MB saved)
  const float* W_emb   = (const float*)d_in[2];
  const float* b_emb   = (const float*)d_in[3];
  const float* W_ih    = (const float*)d_in[4];
  const float* b_ih    = (const float*)d_in[5];
  // d_in[6] W_hh: unused (h0 == 0)
  const float* b_hh    = (const float*)d_in[7];
  const float* query   = (const float*)d_in[8];
  const float* W_attn  = (const float*)d_in[9];
  const float* v_param = (const float*)d_in[10];
  const float* W_val   = (const float*)d_in[11];
  const float* b_val   = (const float*)d_in[12];
  float* out = (float*)d_out;
  float* ws  = (float*)d_ws;   // [0..127]=hx, [128]=v

  lstm_hx_kernel<<<dim3(1), dim3(512), 0, stream>>>(W_ih, b_ih, b_hh, query,
                                                    W_val, b_val, ws);
  attn_kernel<<<dim3(B_), dim3(256), 0, stream>>>(node_features, W_emb, b_emb,
                                                  W_attn, v_param, ws, out);
}
